// DenseGCNBlock_84842783965681
// MI455X (gfx1250) — compile-verified, matrix operand order checked by reading, not run on hardware
//
#include <hip/hip_runtime.h>
#include <hip/hip_bf16.h>

// DenseGCN block, restructured (edge-conv distributes over concat+segment_sum):
//   per layer:  G = X_cat @ Wx              (WMMA f32 dense GEMM, [N,K]@[K,128])
//               ACC[dst] += G[src] + ea@We  (edge scatter, LDS-staged We, atomics)
//               out = ACC/max(cnt,1) + (cnt>0)*b
// Dense outputs h,z1,z2 live contiguously in H[N,384] so layer-l GEMM reads the
// concat for free (row stride 384). Total FLOPs ~7.5G -> memory/atomic bound;
// node arrays are L2-resident (192MB), edge phase ~330MB/layer of L2 traffic.

#define NN       10000
#define IN_CH    128
#define LAYER    128
#define EDGE_DIM 16
#define HSTRIDE  384
#define APAD     4          // LDS row padding: stride K+4 dwords -> conflict-free

typedef __attribute__((ext_vector_type(2)))  float   v2f;
typedef __attribute__((ext_vector_type(8)))  float   v8f;
typedef __attribute__((ext_vector_type(4)))  int     v4i;
typedef __attribute__((ext_vector_type(16))) __bf16  v16bf;

#if defined(__gfx1250__) && __has_builtin(__builtin_amdgcn_wmma_f32_16x16x4_f32)
#define USE_F32_WMMA 1
#else
#define USE_F32_WMMA 0
#endif

#if defined(__gfx1250__) && __has_builtin(__builtin_amdgcn_global_load_async_to_lds_b128)
#define HAVE_ASYNC_LDS 1
typedef __attribute__((address_space(1))) v4i* ag4p;   // global int4*
typedef __attribute__((address_space(3))) v4i* al4p;   // LDS int4*
#else
#define HAVE_ASYNC_LDS 0
#endif

#if defined(__gfx1250__) && __has_builtin(__builtin_amdgcn_s_wait_asynccnt)
#define ASYNC_WAIT() __builtin_amdgcn_s_wait_asynccnt(0)
#else
#define ASYNC_WAIT() asm volatile("s_wait_asynccnt 0x0" ::: "memory")
#endif

// ---------------------------------------------------------------------------
// Dense GEMM: G[N,128] = X[N,K] (row stride ldx) @ W[K,128] (row-major).
// Block = 256 threads = 8 wave32. The 16xK A tile is staged once into LDS
// (async global->LDS when available) and shared by all 8 waves; wave w then
// computes the 16x16 tile at (blockIdx.x*16, w*16) with f32 WMMA over K.
// K in {128,256,384}; N multiple of 16 -> full tiles, EXEC all-ones for WMMA.
// ---------------------------------------------------------------------------
__global__ __launch_bounds__(256)
void gemm_wmma(const float* __restrict__ X, int ldx, int K,
               const float* __restrict__ W, float* __restrict__ G) {
#if defined(__gfx1250__)
  __shared__ float Alds[16 * (3 * LAYER + APAD)];   // up to 16 x (384+4) f32
  const int tid  = threadIdx.x;
  const int lane = tid & 31;
  const int wave = tid >> 5;
  const int row0 = blockIdx.x << 4;
  const int col0 = wave << 4;
  const int mn   = lane & 15;          // A row / B,D col owned by this lane
  const int kh   = lane >> 4;          // which K-half of the fragment
  const int ldA  = K + APAD;
  const int row4 = K >> 2;             // float4 per A row

  // ---- stage A tile: rows handled wave-cyclic, float4 lane-cyclic ----
#if HAVE_ASYNC_LDS
  for (int r = wave; r < 16; r += 8)
    for (int c4 = lane; c4 < row4; c4 += 32)
      __builtin_amdgcn_global_load_async_to_lds_b128(
          (ag4p)(X + (size_t)(row0 + r) * ldx + (c4 << 2)),
          (al4p)(Alds + r * ldA + (c4 << 2)), 0, 0);
  ASYNC_WAIT();
#else
  for (int r = wave; r < 16; r += 8)
    for (int c4 = lane; c4 < row4; c4 += 32)
      *(float4*)(Alds + r * ldA + (c4 << 2)) =
          *(const float4*)(X + (size_t)(row0 + r) * ldx + (c4 << 2));
#endif
  __syncthreads();

  const float* __restrict__ Ar = Alds + mn * ldA;   // ds reads, bank-padded
  v8f acc = {};

#if USE_F32_WMMA
  // A 16x4 f32: lane(kh,m): vgpr0=A[m][2kh], vgpr1=A[m][2kh+1]
  // B 4x16 f32: lane(kh,n): vgpr0=B[2kh][n], vgpr1=B[2kh+1][n]
  for (int k0 = 0; k0 < K; k0 += 4) {
    const int ka = k0 + 2 * kh;
    v2f a, b;
    a.x = Ar[ka];
    a.y = Ar[ka + 1];
    b.x = W[(size_t)ka * LAYER + col0 + mn];
    b.y = W[(size_t)(ka + 1) * LAYER + col0 + mn];
    acc = __builtin_amdgcn_wmma_f32_16x16x4_f32(
        false, a, false, b, (short)0, acc, false, false);
  }
#else
  // Fallback: confirmed bf16 WMMA (16x16x32), f32->bf16 converted operands.
  for (int k0 = 0; k0 < K; k0 += 32) {
    v16bf a, b;
#pragma unroll
    for (int v = 0; v < 8; ++v) {
      const int kk = (v < 4) ? (k0 + 8 * kh + 2 * v)
                             : (k0 + 16 + 8 * kh + 2 * (v - 4));
      a[2 * v]     = (__bf16)Ar[kk];
      a[2 * v + 1] = (__bf16)Ar[kk + 1];
      b[2 * v]     = (__bf16)W[(size_t)kk * LAYER + col0 + mn];
      b[2 * v + 1] = (__bf16)W[(size_t)(kk + 1) * LAYER + col0 + mn];
    }
    acc = __builtin_amdgcn_wmma_f32_16x16x32_bf16(
        false, a, false, b, (short)0, acc, false, false);
  }
#endif

  // D 16x16 f32: vgpr v -> row (v + 8*kh), col = mn
  const int mb = kh << 3;
#pragma unroll
  for (int v = 0; v < 8; ++v)
    G[(size_t)(row0 + mb + v) * LAYER + col0 + mn] = acc[v];
#endif // __gfx1250__
}

// ---------------------------------------------------------------------------
// Edge scatter (persistent grid-stride): one wave32 per edge, lane owns 4 chans.
// We (16x128 = 8KB) is staged in LDS once per block and reused for ~E/waves
// edges, keeping the vector cache path free for the random G[src] gathers and
// ACC[dst] atomics that bound this phase. Next gather row is prefetched.
// ---------------------------------------------------------------------------
__global__ __launch_bounds__(256)
void edge_scatter(const float* __restrict__ G, const float* __restrict__ EA,
                  const int* __restrict__ src, const int* __restrict__ dst,
                  const float* __restrict__ We, float* __restrict__ ACC,
                  int nE) {
  __shared__ float Wlds[EDGE_DIM * LAYER];
  const int tid = threadIdx.x;
  for (int i = tid; i < EDGE_DIM * LAYER; i += 256) Wlds[i] = We[i];
  __syncthreads();

  const int lane   = tid & 31;
  const int c      = lane << 2;                       // 4 channels per lane
  const int stride = (gridDim.x * blockDim.x) >> 5;   // total waves

  for (int e = (blockIdx.x * (blockDim.x >> 5)) + (tid >> 5); e < nE;
       e += stride) {
    const int en = e + stride;                        // software pipeline:
    if (en < nE)                                      // prefetch next gather row
      __builtin_prefetch(G + (size_t)src[en] * LAYER + c, 0, 0);

    const int s = src[e];
    const int d = dst[e];
    const float4 g = *(const float4*)(G + (size_t)s * LAYER + c);
    const float4* __restrict__ ea4 = (const float4*)(EA + (size_t)e * EDGE_DIM);
    float px = g.x, py = g.y, pz = g.z, pw = g.w;
#pragma unroll
    for (int q = 0; q < 4; ++q) {
      const float4 ev = ea4[q];
      const float* wl = Wlds + (q << 2) * LAYER + c;  // ds_load_b128 x4
      const float4 w0 = *(const float4*)(wl);
      const float4 w1 = *(const float4*)(wl + LAYER);
      const float4 w2 = *(const float4*)(wl + 2 * LAYER);
      const float4 w3 = *(const float4*)(wl + 3 * LAYER);
      px += ev.x * w0.x + ev.y * w1.x + ev.z * w2.x + ev.w * w3.x;
      py += ev.x * w0.y + ev.y * w1.y + ev.z * w2.y + ev.w * w3.y;
      pz += ev.x * w0.z + ev.y * w1.z + ev.z * w2.z + ev.w * w3.z;
      pw += ev.x * w0.w + ev.y * w1.w + ev.z * w2.w + ev.w * w3.w;
    }
    float* a = ACC + (size_t)d * LAYER + c;
    atomicAdd(a + 0, px);
    atomicAdd(a + 1, py);
    atomicAdd(a + 2, pz);
    atomicAdd(a + 3, pw);
  }
}

// ---------------------------------------------------------------------------
__global__ void zero_f32(float* __restrict__ p, int n) {
  const int i = blockIdx.x * blockDim.x + threadIdx.x;
  if (i < n) p[i] = 0.f;
}

__global__ void count_k(const int* __restrict__ dst, float* __restrict__ cnt,
                        int nE) {
  const int i = blockIdx.x * blockDim.x + threadIdx.x;
  if (i < nE) atomicAdd(&cnt[dst[i]], 1.0f);
}

// out[n, ldo] = ACC/max(cnt,1) + (cnt>0)*bias
__global__ void finalize_k(const float* __restrict__ ACC,
                           const float* __restrict__ counts,
                           const float* __restrict__ bias,
                           float* __restrict__ O, int ldo, int N) {
  const int i = blockIdx.x * blockDim.x + threadIdx.x;
  if (i >= N * LAYER) return;
  const int n = i >> 7;
  const int c = i & (LAYER - 1);
  const float cnt = counts[n];
  const float v = ACC[i] / fmaxf(cnt, 1.0f) + (cnt > 0.f ? bias[c] : 0.f);
  O[(size_t)n * ldo + c] = v;
}

// ---------------------------------------------------------------------------
extern "C" void kernel_launch(void* const* d_in, const int* in_sizes, int n_in,
                              void* d_out, int out_size, void* d_ws,
                              size_t ws_size, hipStream_t stream) {
  const float* x  = (const float*)d_in[0];
  const float* ea = (const float*)d_in[1];
  const int*   ei = (const int*)d_in[2];
  const float* W[4] = {(const float*)d_in[3], (const float*)d_in[5],
                       (const float*)d_in[7], (const float*)d_in[9]};
  const float* B[4] = {(const float*)d_in[4], (const float*)d_in[6],
                       (const float*)d_in[8], (const float*)d_in[10]};

  const int N = in_sizes[0] / IN_CH;   // 10000 (multiple of 16)
  const int E = in_sizes[2] / 2;       // 320000
  const int* src = ei;
  const int* dst = ei + E;

  // Workspace layout (floats): counts[N] | H[N*384] | G[N*128] | ACC[N*128]
  float* ws     = (float*)d_ws;
  float* counts = ws;
  float* H      = ws + N;
  float* G      = H + (size_t)N * HSTRIDE;
  float* ACC    = G + (size_t)N * LAYER;
  float* out    = (float*)d_out;

  const int NC = N * LAYER;
  int nEdgeBlk = (E + 7) / 8;
  if (nEdgeBlk > 2048) nEdgeBlk = 2048;   // persistent edge blocks

  zero_f32<<<(N + 255) / 256, 256, 0, stream>>>(counts, N);
  count_k<<<(E + 255) / 256, 256, 0, stream>>>(dst, counts, E);

  const int Ks[4] = {IN_CH, LAYER, 2 * LAYER, 3 * LAYER};
  for (int l = 0; l < 4; ++l) {
    const float* Xl  = (l == 0) ? x : H;
    const int    ldx = (l == 0) ? IN_CH : HSTRIDE;
    const int    K   = Ks[l];
    const float* Wx  = W[l];                       // first K rows
    const float* We  = W[l] + (size_t)K * LAYER;   // last 16 rows

    gemm_wmma<<<N / 16, 256, 0, stream>>>(Xl, ldx, K, Wx, G);
    zero_f32<<<(NC + 255) / 256, 256, 0, stream>>>(ACC, NC);
    edge_scatter<<<nEdgeBlk, 256, 0, stream>>>(G, ea, src, dst, We, ACC, E);

    float* O   = (l == 3) ? out : (H + (size_t)l * LAYER);
    int    ldo = (l == 3) ? LAYER : HSTRIDE;
    finalize_k<<<(NC + 255) / 256, 256, 0, stream>>>(ACC, counts, B[l], O, ldo, N);
  }
}
